// PGAgent_ti_87668872446278
// MI455X (gfx1250) — compile-verified, matrix operand order checked by reading, not run on hardware
//
#include <hip/hip_runtime.h>
#include <hip/hip_bf16.h>

#define BATCH   1024
#define HID     2048
#define G4      8192     // 4*HID
#define NTAKE   64
#define NBLOCK  128
#define NYARD   10
#define INPK    80
#define INPKP   96       // padded to multiple of 32
#define LIMIT_V 640.0f
#define BETA_V  0.1f

#define KTILE      32
#define LDS_STRIDE 40                    // 32 K elems + 8 pad (80B rows, 16B aligned)
#define TILE_ELEMS (128 * LDS_STRIDE)    // 5120 bf16 per tile

typedef __bf16 bf16_t;
typedef bf16_t bfx8  __attribute__((ext_vector_type(8)));
typedef bf16_t v16bf __attribute__((ext_vector_type(16)));
typedef float  v8f   __attribute__((ext_vector_type(8)));

// ---------------------------------------------------------------------------
// CDNA5 async global->LDS copy (ASYNCcnt-tracked) + wait.
// ---------------------------------------------------------------------------
__device__ __forceinline__ void async_copy_b128(unsigned lds_byte_off, const void* gptr) {
    asm volatile("global_load_async_to_lds_b128 %0, %1, off"
                 :: "v"(lds_byte_off), "v"(gptr) : "memory");
}
__device__ __forceinline__ void wait_async0() {
    asm volatile("s_wait_asynccnt 0x0" ::: "memory");
}

// ---------------------------------------------------------------------------
// WMMA fragment loads.  Per ISA 16-bit A/B layout: lane l (r=l&15, hi=l>>4)
// holds two contiguous 8-element K runs: [k0+8*hi, +8) and [k0+16+8*hi, +8).
// ---------------------------------------------------------------------------
__device__ __forceinline__ v16bf load_frag(const bf16_t* __restrict__ base,
                                           int row0, int ld, int k0, int lane) {
    const int r  = lane & 15;
    const int hi = lane >> 4;
    const bf16_t* p = base + (size_t)(row0 + r) * (size_t)ld + (size_t)(k0 + hi * 8);
    bfx8 lo = *(const bfx8*)(p);
    bfx8 hh = *(const bfx8*)(p + 16);
    return __builtin_shufflevector(lo, hh, 0,1,2,3,4,5,6,7,8,9,10,11,12,13,14,15);
}

// Fragment gather from a staged LDS tile (row-major, LDS_STRIDE elems/row).
__device__ __forceinline__ v16bf lds_frag(const bf16_t* sbase, int subrow0, int lane) {
    const int r  = lane & 15;
    const int hi = lane >> 4;
    const bf16_t* p = sbase + (subrow0 + r) * LDS_STRIDE + hi * 8;
    bfx8 lo = *(const bfx8*)(p);
    bfx8 hh = *(const bfx8*)(p + 16);
    return __builtin_shufflevector(lo, hh, 0,1,2,3,4,5,6,7,8,9,10,11,12,13,14,15);
}

__device__ __forceinline__ v8f wmma_bf16(v16bf a, v16bf b, v8f c) {
    return __builtin_amdgcn_wmma_f32_16x16x32_bf16(false, a, false, b,
                                                   (short)0, c, false, false);
}

__device__ __forceinline__ float sigm(float x) { return 1.0f / (1.0f + __expf(-x)); }

__device__ __forceinline__ unsigned int hash_u32(unsigned int x) {
    x ^= x >> 16; x *= 0x7feb352du;
    x ^= x >> 15; x *= 0x846ca68bu;
    x ^= x >> 16; return x;
}

// ---------------------------------------------------------------------------
// gates = inp_bf(1024x96) @ W_ih_padT + h_bf(1024x2048) @ W_hhT + bsum
// block: 256 thr = 8 waves; tile 128(M) x 128(N); grid (8, 64).
// Phase 2 stages A(128x32) + B(128x32) bf16 tiles in LDS, double-buffered,
// filled by async global->LDS b128 copies.  Wave w computes 4(M) x 2(N)
// 16x16 subtiles: m-subtiles (w&1)*4+{0..3}, n-subtiles (w>>1)*2+{0,1}.
// ---------------------------------------------------------------------------
__global__ __launch_bounds__(256)
void gates_gemm_kernel(const bf16_t* __restrict__ inp_bf,
                       const bf16_t* __restrict__ w_ih_bf,   // 8192 x 96
                       const bf16_t* __restrict__ h_bf,      // 1024 x 2048
                       const bf16_t* __restrict__ w_hh_bf,   // 8192 x 2048
                       const float*  __restrict__ bsum,      // 8192
                       float*        __restrict__ gates) {   // 1024 x 8192
    __shared__ bf16_t smem[4 * TILE_ELEMS];   // [buf][A|B] : 40 KB

    const int tid  = threadIdx.x;
    const int wave = tid >> 5;
    const int lane = tid & 31;
    const int m_base = blockIdx.x * 128;
    const int n_base = blockIdx.y * 128;
    const int ms0 = (wave & 1) * 4;     // first of 4 m-subtiles (16 rows each)
    const int ns0 = (wave >> 1) * 2;    // first of 2 n-subtiles

    v8f acc[4][2];
    #pragma unroll
    for (int i = 0; i < 4; ++i)
        #pragma unroll
        for (int j = 0; j < 2; ++j) acc[i][j] = (v8f){};

    // ---- phase 1: input projection, K = 96 (padded 80), direct loads ----
    #pragma unroll
    for (int k = 0; k < INPKP; k += KTILE) {
        v16bf b0 = load_frag(w_ih_bf, n_base + (ns0 + 0) * 16, INPKP, k, lane);
        v16bf b1 = load_frag(w_ih_bf, n_base + (ns0 + 1) * 16, INPKP, k, lane);
        #pragma unroll
        for (int i = 0; i < 4; ++i) {
            v16bf a = load_frag(inp_bf, m_base + (ms0 + i) * 16, INPKP, k, lane);
            acc[i][0] = wmma_bf16(a, b0, acc[i][0]);
            acc[i][1] = wmma_bf16(a, b1, acc[i][1]);
        }
    }

    // ---- phase 2: recurrent projection, K = 2048, LDS double-buffered ----
    const unsigned lds_base = (unsigned)(size_t)(&smem[0]);  // LDS byte address
    // stage one 128x32 tile: 512 b128 chunks, 2 per thread
    auto stage = [&](const bf16_t* gbase, int row0, int kglob, int buf, int which) {
        const unsigned tbase = lds_base + (unsigned)((buf * 2 + which) * TILE_ELEMS) * 2u;
        #pragma unroll
        for (int i = 0; i < 2; ++i) {
            const int c   = tid + i * 256;     // chunk id < 512
            const int row = c >> 2;
            const int ko  = (c & 3) * 8;
            const bf16_t* g = gbase + (size_t)(row0 + row) * HID + (size_t)(kglob + ko);
            async_copy_b128(tbase + (unsigned)(row * LDS_STRIDE + ko) * 2u, (const void*)g);
        }
    };

    stage(h_bf,    m_base, 0, 0, 0);
    stage(w_hh_bf, n_base, 0, 0, 1);

    const int NK = HID / KTILE;   // 64
    for (int kt = 0; kt < NK; ++kt) {
        const int buf = kt & 1;
        wait_async0();
        __syncthreads();                       // tile `buf` visible to all waves
        if (kt + 1 < NK) {
            stage(h_bf,    m_base, (kt + 1) * KTILE, buf ^ 1, 0);
            stage(w_hh_bf, n_base, (kt + 1) * KTILE, buf ^ 1, 1);
        }
        const bf16_t* sA = &smem[(buf * 2 + 0) * TILE_ELEMS];
        const bf16_t* sB = &smem[(buf * 2 + 1) * TILE_ELEMS];
        v16bf b0 = lds_frag(sB, (ns0 + 0) * 16, lane);
        v16bf b1 = lds_frag(sB, (ns0 + 1) * 16, lane);
        #pragma unroll
        for (int i = 0; i < 4; ++i) {
            v16bf a = lds_frag(sA, (ms0 + i) * 16, lane);
            acc[i][0] = wmma_bf16(a, b0, acc[i][0]);
            acc[i][1] = wmma_bf16(a, b1, acc[i][1]);
        }
        __syncthreads();                       // reads done before buf is re-filled
    }

    // ---- epilogue: lane -> col n+(l&15); VGPR v -> row m + v + 8*(l>>4) ----
    const int r = lane & 15, hi = lane >> 4;
    #pragma unroll
    for (int i = 0; i < 4; ++i) {
        const int m0 = m_base + (ms0 + i) * 16;
        #pragma unroll
        for (int j = 0; j < 2; ++j) {
            const int n = n_base + (ns0 + j) * 16 + r;
            const float bv = bsum[n];
            #pragma unroll
            for (int v = 0; v < 8; ++v)
                gates[(size_t)(m0 + v + 8 * hi) * G4 + n] = acc[i][j][v] + bv;
        }
    }
}

// ---------------------------------------------------------------------------
// logits = h_bf(1024x2048) @ fc_wT(2048x128) + fc_b
// block: 256 thr = 8 waves; M tile 16; wave w owns columns [16w, 16w+16)
// grid: 64
// ---------------------------------------------------------------------------
__global__ __launch_bounds__(256)
void logits_gemm_kernel(const bf16_t* __restrict__ h_bf,
                        const bf16_t* __restrict__ fc_w_bf,  // 128 x 2048
                        const float*  __restrict__ fc_b,     // 128
                        float*        __restrict__ logits) { // 1024 x 128
    const int wave = threadIdx.x >> 5;
    const int lane = threadIdx.x & 31;
    const int m0 = blockIdx.x * 16;
    const int n0 = wave * 16;

    v8f acc = {};
    #pragma unroll 4
    for (int k = 0; k < HID; k += 32) {
        v16bf a = load_frag(h_bf,    m0, HID, k, lane);
        v16bf b = load_frag(fc_w_bf, n0, HID, k, lane);
        acc = wmma_bf16(a, b, acc);
    }
    const int r = lane & 15, hi = lane >> 4;
    const float bv = fc_b[n0 + r];
    #pragma unroll
    for (int v = 0; v < 8; ++v)
        logits[(size_t)(m0 + v + 8 * hi) * NBLOCK + n0 + r] = acc[v] + bv;
}

// ---------------------------------------------------------------------------
// LSTM cell elementwise.  gates order: i, f, g, o (torch LSTMCell).
// ---------------------------------------------------------------------------
__global__ __launch_bounds__(256)
void lstm_cell_kernel(const float* __restrict__ gates,
                      float* __restrict__ c, float* __restrict__ h,
                      bf16_t* __restrict__ h_bf) {
    const int idx = blockIdx.x * blockDim.x + threadIdx.x;   // < 1024*2048
    const int b = idx >> 11;
    const int j = idx & (HID - 1);
    const float* g = gates + (size_t)b * G4;
    const float ig = g[j], fg = g[HID + j], gg = g[2 * HID + j], og = g[3 * HID + j];
    const float cc = sigm(fg) * c[idx] + sigm(ig) * tanhf(gg);
    const float hh = sigm(og) * tanhf(cc);
    c[idx] = cc;
    h[idx] = hh;
    h_bf[idx] = (bf16_t)hh;
}

// ---------------------------------------------------------------------------
// Per-row: mask logits, Gumbel-max sample, log-softmax logp, bookkeeping,
// build next LSTM input.  block = 128 threads (one per logit), grid = 1024.
// ---------------------------------------------------------------------------
__global__ __launch_bounds__(128)
void sample_step_kernel(const float* __restrict__ logits,
                        const float* __restrict__ take_in,      // 64 x 5
                        const int*   __restrict__ take_in_loc,  // 64
                        const float* __restrict__ yard_slots,   // 10 x 15 x 5
                        int*   __restrict__ bmask,              // 1024 x 64
                        float* __restrict__ count,              // 1024 x 10
                        float* __restrict__ rem,                // 1024 x 10
                        bf16_t* __restrict__ inp_bf,            // 1024 x 96
                        float* __restrict__ out, int t) {
    __shared__ float sml[NBLOCK];
    __shared__ float red[NBLOCK];
    __shared__ int   redi[NBLOCK];
    __shared__ int   s_action;

    const int b = blockIdx.x;
    const int j = threadIdx.x;

    float l = logits[(size_t)b * NBLOCK + j];
    const bool masked = (j >= NTAKE) || (bmask[b * NTAKE + j] != 0);
    l = masked ? -20000.0f : l;
    sml[j] = l;

    // Gumbel noise from a counter hash keyed on (t, b, j)
    unsigned int s = hash_u32(((unsigned)t * 1024u + (unsigned)b) * 131u +
                              (unsigned)j + 0x9e3779b9u);
    float u = (float)(s >> 8) * (1.0f / 16777216.0f) + 1e-7f;
    float gum = -__logf(-__logf(u));
    red[j]  = l + gum;
    redi[j] = j;
    __syncthreads();
    for (int off = NBLOCK / 2; off > 0; off >>= 1) {
        if (j < off && red[j + off] > red[j]) { red[j] = red[j + off]; redi[j] = redi[j + off]; }
        __syncthreads();
    }
    if (j == 0) s_action = redi[0];
    __syncthreads();
    const int action = s_action;

    // max of masked logits
    red[j] = sml[j];
    __syncthreads();
    for (int off = NBLOCK / 2; off > 0; off >>= 1) {
        if (j < off) red[j] = fmaxf(red[j], red[j + off]);
        __syncthreads();
    }
    const float mx = red[0];
    __syncthreads();
    // sum exp
    red[j] = __expf(sml[j] - mx);
    __syncthreads();
    for (int off = NBLOCK / 2; off > 0; off >>= 1) {
        if (j < off) red[j] += red[j + off];
        __syncthreads();
    }

    const int yard = take_in_loc[action];

    if (j == 0) {
        const float logp = sml[action] - mx - __logf(red[0]);
        const float area = take_in[action * 5 + 0] * take_in[action * 5 + 1];
        float* cnt = count + b * NYARD;
        float* rm  = rem   + b * NYARD;
        float cy = cnt[yard];
        if (cy == 0.0f) cy = 1.0f;                 // first use of this yard
        const float ry = rm[yard] - area;
        if (ry < 0.0f) cy += 1.0f;                 // yard overflow -> new batch
        cnt[yard] = cy;
        rm[yard]  = ry;
        float sc = 0.0f;
        #pragma unroll
        for (int y = 0; y < NYARD; ++y) sc += cnt[y];
        bmask[b * NTAKE + action] = 1;
        out[(size_t)b * NTAKE + t]                         = logp;           // logps
        out[(size_t)BATCH * NTAKE + (size_t)b * NTAKE + t] = -BETA_V * sc;   // rewards
        out[2ull * BATCH * NTAKE + (size_t)b * NTAKE + t]  = (float)action;  // actions
    }

    // build next input: [block(5) | yard_slots[yard] (75)] padded to 96 bf16
    if (j < INPK) {
        float v;
        if (j < 5)
            v = (j == 4) ? (float)(yard + 1) : take_in[action * 5 + j];
        else
            v = yard_slots[(yard * 15 + (j - 5) / 5) * 5 + (j - 5) % 5];
        inp_bf[(size_t)b * INPKP + j] = (bf16_t)v;
    } else if (j < INPKP) {
        inp_bf[(size_t)b * INPKP + j] = (bf16_t)0.0f;
    }
}

// ---------------------------------------------------------------------------
// init / conversion helpers
// ---------------------------------------------------------------------------
__global__ void f2bf_kernel(const float* __restrict__ s, bf16_t* __restrict__ d, int n) {
    int i = blockIdx.x * blockDim.x + threadIdx.x;
    if (i < n) d[i] = (bf16_t)s[i];
}
__global__ void pack_wih_kernel(const float* __restrict__ w, bf16_t* __restrict__ d) {
    int i = blockIdx.x * blockDim.x + threadIdx.x;       // < 8192*96
    if (i < G4 * INPKP) {
        int r = i / INPKP, col = i % INPKP;
        d[i] = (col < INPK) ? (bf16_t)w[r * INPK + col] : (bf16_t)0.0f;
    }
}
__global__ void fill_f32_kernel(float* __restrict__ p, float v, int n) {
    int i = blockIdx.x * blockDim.x + threadIdx.x;
    if (i < n) p[i] = v;
}
__global__ void fill_i32_kernel(int* __restrict__ p, int v, int n) {
    int i = blockIdx.x * blockDim.x + threadIdx.x;
    if (i < n) p[i] = v;
}
__global__ void fill_bf_kernel(bf16_t* __restrict__ p, int n) {
    int i = blockIdx.x * blockDim.x + threadIdx.x;
    if (i < n) p[i] = (bf16_t)0.0f;
}
__global__ void vec_add_kernel(const float* __restrict__ a, const float* __restrict__ b,
                               float* __restrict__ o, int n) {
    int i = blockIdx.x * blockDim.x + threadIdx.x;
    if (i < n) o[i] = a[i] + b[i];
}

// ---------------------------------------------------------------------------
extern "C" void kernel_launch(void* const* d_in, const int* in_sizes, int n_in,
                              void* d_out, int out_size, void* d_ws, size_t ws_size,
                              hipStream_t stream) {
    const float* take_in     = (const float*)d_in[0];
    const int*   take_in_loc = (const int*)  d_in[1];
    const float* yard_slots  = (const float*)d_in[2];
    const float* W_ih        = (const float*)d_in[3];
    const float* W_hh        = (const float*)d_in[4];
    const float* b_ih        = (const float*)d_in[5];
    const float* b_hh        = (const float*)d_in[6];
    const float* fc_w        = (const float*)d_in[7];
    const float* fc_b        = (const float*)d_in[8];
    float* out = (float*)d_out;

    char* ws = (char*)d_ws;
    auto alloc = [&](size_t bytes) -> char* {
        char* p = ws; ws += (bytes + 255) & ~(size_t)255; return p;
    };
    bf16_t* w_hh_bf = (bf16_t*)alloc((size_t)G4 * HID * 2);
    bf16_t* w_ih_bf = (bf16_t*)alloc((size_t)G4 * INPKP * 2);
    bf16_t* fc_w_bf = (bf16_t*)alloc((size_t)NBLOCK * HID * 2);
    float*  bsum    = (float*) alloc((size_t)G4 * 4);
    float*  gates   = (float*) alloc((size_t)BATCH * G4 * 4);
    float*  h       = (float*) alloc((size_t)BATCH * HID * 4);
    float*  c       = (float*) alloc((size_t)BATCH * HID * 4);
    bf16_t* h_bf    = (bf16_t*)alloc((size_t)BATCH * HID * 2);
    float*  logits  = (float*) alloc((size_t)BATCH * NBLOCK * 4);
    bf16_t* inp_bf  = (bf16_t*)alloc((size_t)BATCH * INPKP * 2);
    float*  count   = (float*) alloc((size_t)BATCH * NYARD * 4);
    float*  rem     = (float*) alloc((size_t)BATCH * NYARD * 4);
    int*    bmask   = (int*)   alloc((size_t)BATCH * NTAKE * 4);

    const int T = 256;
    auto blks = [](int n, int t) { return (n + t - 1) / t; };

    // one-time conversions + state init
    f2bf_kernel<<<blks(G4 * HID, T), T, 0, stream>>>(W_hh, w_hh_bf, G4 * HID);
    pack_wih_kernel<<<blks(G4 * INPKP, T), T, 0, stream>>>(W_ih, w_ih_bf);
    f2bf_kernel<<<blks(NBLOCK * HID, T), T, 0, stream>>>(fc_w, fc_w_bf, NBLOCK * HID);
    vec_add_kernel<<<blks(G4, T), T, 0, stream>>>(b_ih, b_hh, bsum, G4);
    fill_f32_kernel<<<blks(BATCH * HID, T), T, 0, stream>>>(h, 0.0f, BATCH * HID);
    fill_f32_kernel<<<blks(BATCH * HID, T), T, 0, stream>>>(c, 0.0f, BATCH * HID);
    fill_bf_kernel<<<blks(BATCH * HID, T), T, 0, stream>>>(h_bf, BATCH * HID);
    fill_bf_kernel<<<blks(BATCH * INPKP, T), T, 0, stream>>>(inp_bf, BATCH * INPKP);
    fill_f32_kernel<<<blks(BATCH * NYARD, T), T, 0, stream>>>(count, 0.0f, BATCH * NYARD);
    fill_f32_kernel<<<blks(BATCH * NYARD, T), T, 0, stream>>>(rem, LIMIT_V, BATCH * NYARD);
    fill_i32_kernel<<<blks(BATCH * NTAKE, T), T, 0, stream>>>(bmask, 0, BATCH * NTAKE);

    // 64 serialized decode steps
    for (int t = 0; t < NTAKE; ++t) {
        gates_gemm_kernel<<<dim3(BATCH / 128, G4 / 128), 256, 0, stream>>>(
            inp_bf, w_ih_bf, h_bf, w_hh_bf, bsum, gates);
        lstm_cell_kernel<<<blks(BATCH * HID, T), T, 0, stream>>>(gates, c, h, h_bf);
        logits_gemm_kernel<<<BATCH / 16, 256, 0, stream>>>(h_bf, fc_w_bf, fc_b, logits);
        sample_step_kernel<<<BATCH, NBLOCK, 0, stream>>>(
            logits, take_in, take_in_loc, yard_slots, bmask, count, rem, inp_bf, out, t);
    }
}
